// ODERNN_52913997087345
// MI455X (gfx1250) — matrix operand order repaired and checked by reading, not compiled, not run on hardware
//
#include <hip/hip_runtime.h>
#include <math.h>

// ---------------------------------------------------------------------------
// ODE-RNN forward on gfx1250 (MI455X).
// B=1024, T=100, D=32, H=256, MID=50(pad 64), RK4 x 4 steps.
// 64 workgroups x 128 threads (4 waves). Each WG owns 16 batch rows for all T.
// Hidden state kept in registers (f32, D-frag layout) + f16 LDS mirror (A mat).
// All GEMMs via v_wmma_f32_16x16x32_f16.
// ---------------------------------------------------------------------------

typedef __attribute__((ext_vector_type(16))) _Float16 v16h;
typedef __attribute__((ext_vector_type(8)))  float    v8f;
typedef __attribute__((ext_vector_type(2)))  __fp16   fp16x2;  // cvt_pkrtz result type

union Frag { v16h h; unsigned int u[8]; };

#define Bsz 1024
#define Tsz 100
#define Dsz 32
#define Hsz 256
#define MIDP 64   // ODE mid width padded 50 -> 64

__device__ inline unsigned pack2(float a, float b) {
  fp16x2 r = __builtin_amdgcn_cvt_pkrtz(a, b);   // v_cvt_pk_rtz_f16_f32
  return __builtin_bit_cast(unsigned, r);
}

// A-matrix fragment (16xK slab, row-major f16 in LDS).
// Lane 0-15: M=lane, K-halves 0..15 block; lane 16-31: same M, upper halves.
// VGPR v: k = k0 + 2*v + 16*hi   (contiguous k order, matches B loaders)
__device__ inline v16h load_a(const _Float16* base, int row, int k0, int stride, int hi) {
  Frag f;
#pragma unroll
  for (int v = 0; v < 8; ++v) {
    int k = k0 + 2 * v + 16 * hi;
    f.u[v] = *(const unsigned*)(base + row * stride + k);
  }
  return f.h;
}

// B-matrix fragment from LDS stored [n][k] f16 (column n contiguous in k).
// Lane column n = nb + (lane&15); VGPR v: k = k0 + 2v + 16*hi.
__device__ inline v16h load_b_lds(const _Float16* base, int n, int k0, int kstride, int hi) {
  Frag f;
#pragma unroll
  for (int v = 0; v < 8; ++v) {
    int k = k0 + 2 * v + 16 * hi;
    f.u[v] = *(const unsigned*)(base + n * kstride + k);
  }
  return f.h;
}

// B-matrix fragment streamed from global f32 weight stored [n][k] row-major.
__device__ inline v16h load_b_g32(const float* base, int n, int k0, int kstride, int hi) {
  Frag f;
#pragma unroll
  for (int v = 0; v < 8; ++v) {
    int k = k0 + 2 * v + 16 * hi;
    const float2 t = *(const float2*)(base + n * kstride + k);  // global_load_b64
    f.u[v] = pack2(t.x, t.y);
  }
  return f.h;
}

__device__ inline v8f wmma_f16(v16h a, v16h b, v8f c) {
  return __builtin_amdgcn_wmma_f32_16x16x32_f16(false, a, false, b, (short)0, c,
                                                false, false);
}

// LDS halves layout (dynamic LDS, ~219 KB):
#define OFF_WHH 0                       // [256][256] f16  (B for h @ W_hh^T)
#define OFF_WIH (OFF_WHH + 65536)       // [256][32]  f16
#define OFF_W1  (OFF_WIH + 8192)        // [64][256]  f16  (mid n, k over H; rows>=50 zero)
#define OFF_W2  (OFF_W1  + 16384)       // [256][64]  f16  (k over mid; k>=50 zero)
#define OFF_HA  (OFF_W2  + 16384)       // [16][256]  f16  current y / h  (A matrix)
#define OFF_ZA  (OFF_HA  + 4096)        // [16][64]   f16  ODE mid activations
#define OFF_XA  (OFF_ZA  + 1024)        // [16][32]   f16  x tile
#define HALVES  (OFF_XA + 512)
#define SMEM_BYTES (HALVES * 2 + 48 * 4)  // + sc[16], mu[16], sg[16] f32

__global__ __launch_bounds__(128) void odernn_fused_kernel(
    const float* __restrict__ dt,     const float* __restrict__ x,
    const float* __restrict__ W_ih,   const float* __restrict__ b_ih,
    const float* __restrict__ W_hh,   const float* __restrict__ b_hh,
    const float* __restrict__ ode_w1, const float* __restrict__ ode_b1,
    const float* __restrict__ ode_w2, const float* __restrict__ ode_b2,
    const float* __restrict__ l1_w,   const float* __restrict__ l1_b,
    const float* __restrict__ mu_w,   const float* __restrict__ mu_b,
    const float* __restrict__ sig_w1, const float* __restrict__ sig_b1,
    const float* __restrict__ sig_w2, const float* __restrict__ sig_b2,
    float* __restrict__ out) {
  extern __shared__ char smem[];
  _Float16* sH  = (_Float16*)smem;
  _Float16* sWhh = sH + OFF_WHH;
  _Float16* sWih = sH + OFF_WIH;
  _Float16* sW1  = sH + OFF_W1;
  _Float16* sW2  = sH + OFF_W2;
  _Float16* sHA  = sH + OFF_HA;
  _Float16* sZA  = sH + OFF_ZA;
  _Float16* sXA  = sH + OFF_XA;
  float* sSC = (float*)(sH + HALVES);
  float* sMU = sSC + 16;
  float* sSG = sMU + 16;

  const int tid  = threadIdx.x;
  const int wave = tid >> 5;       // 0..3
  const int lane = tid & 31;
  const int c    = lane & 15;      // fragment column / A row
  const int hi   = lane >> 4;      // lane-half
  const int row0 = blockIdx.x * 16;

  // ---- Stage weights into LDS as f16 (straight row-major copies) ----
  for (int i = tid; i < Hsz * Hsz; i += 128) sWhh[i] = (_Float16)W_hh[i];
  for (int i = tid; i < Hsz * Dsz; i += 128) sWih[i] = (_Float16)W_ih[i];
  for (int i = tid; i < MIDP * Hsz; i += 128) {
    int mid = i >> 8, k = i & 255;
    sW1[i] = (_Float16)((mid < 50) ? ode_w1[mid * Hsz + k] : 0.f);
  }
  for (int i = tid; i < Hsz * MIDP; i += 128) {
    int n = i >> 6, k = i & 63;
    sW2[i] = (_Float16)((k < 50) ? ode_w2[n * 50 + k] : 0.f);
  }
  for (int i = tid; i < 16 * Hsz; i += 128) sHA[i] = (_Float16)0.f;

  // ---- Per-lane bias / head-weight preloads (D-frag column ownership) ----
  float b1v = 0.f;
  { int n = 16 * wave + c; if (n < 50) b1v = ode_b1[n]; }
  float b2v[4], brnn[4], sb1[4], sw2[4];
#pragma unroll
  for (int nt = 0; nt < 4; ++nt) {
    int n = 64 * wave + 16 * nt + c;
    b2v[nt]  = ode_b2[n];
    brnn[nt] = b_ih[n] + b_hh[n];
    sb1[nt]  = sig_b1[n];
    sw2[nt]  = sig_w2[n];
  }
  float l1bv[2], muwv[2];
#pragma unroll
  for (int tt = 0; tt < 2; ++tt) {
    int n = 32 * wave + 16 * tt + c;
    l1bv[tt] = l1_b[n];
    muwv[tt] = mu_w[n];
  }
  const float mub = mu_b[0], sgb = sig_b2[0];

  // Hidden state, distributed in D-frag layout: hreg[nt][j] = h[j+8*hi][64w+16nt+c]
  v8f hreg[4];
#pragma unroll
  for (int nt = 0; nt < 4; ++nt)
#pragma unroll
    for (int j = 0; j < 8; ++j) hreg[nt][j] = 0.f;

  __syncthreads();

  const float hsv = 0.25f;                       // RK4 sub-step
  const float cnext[3] = {0.5f * hsv, 0.5f * hsv, hsv};
  const float wgt[4]   = {1.f, 2.f, 2.f, 1.f};

  for (int t = 0; t < Tsz; ++t) {
    // ---- Stage x tile + per-row dt scale; reset head accumulators ----
#pragma unroll
    for (int e = 0; e < 4; ++e) {
      int i = tid * 4 + e, r = i >> 5, col = i & 31;
      sXA[i] = (_Float16)x[((row0 + r) * Tsz + t) * Dsz + col];
    }
    if (tid < 16) {
      int b = row0 + tid;
      float d0 = dt[(b * Tsz + t) * 2 + 0];
      float d1 = dt[(b * Tsz + t) * 2 + 1];
      sSC[tid] = (d1 - d0) * (1.f / 24.f);
      sMU[tid] = mub;
      sSG[tid] = sgb;
    }
    __syncthreads();

    float scv[8];                                 // dt scale per owned row
#pragma unroll
    for (int j = 0; j < 8; ++j) scv[j] = sSC[j + 8 * hi];

    // ================= RNN cell: h += tanh(x W_ih^T + h W_hh^T + b) ========
    v8f pre[4];
#pragma unroll
    for (int nt = 0; nt < 4; ++nt) {
      int nb = 64 * wave + 16 * nt;
      v8f acc;
#pragma unroll
      for (int j = 0; j < 8; ++j) acc[j] = brnn[nt];
      acc = wmma_f16(load_a(sXA, c, 0, Dsz, hi),
                     load_b_lds(sWih, nb + c, 0, Dsz, hi), acc);
#pragma unroll
      for (int k0 = 0; k0 < Hsz; k0 += 32)
        acc = wmma_f16(load_a(sHA, c, k0, Hsz, hi),
                       load_b_lds(sWhh, nb + c, k0, Hsz, hi), acc);
      pre[nt] = acc;
    }
    __syncthreads();                              // all sHA reads done
#pragma unroll
    for (int nt = 0; nt < 4; ++nt) {
      int nb = 64 * wave + 16 * nt;
#pragma unroll
      for (int j = 0; j < 8; ++j) {
        hreg[nt][j] += tanhf(pre[nt][j]);
        sHA[(j + 8 * hi) * Hsz + nb + c] = (_Float16)hreg[nt][j];
      }
    }
    __syncthreads();

    // ================= RK4: dh/dt = (tanh(y W1^T + b1) W2^T + b2) * scale ==
    v8f ksum[4];
#pragma unroll
    for (int nt = 0; nt < 4; ++nt)
#pragma unroll
      for (int j = 0; j < 8; ++j) ksum[nt][j] = 0.f;

    for (int s = 0; s < 4; ++s) {
      // GEMM1: mid = tanh(y @ W1^T + b1); this wave owns mid cols [16w,16w+16)
      v8f z;
#pragma unroll
      for (int j = 0; j < 8; ++j) z[j] = b1v;
#pragma unroll
      for (int k0 = 0; k0 < Hsz; k0 += 32)
        z = wmma_f16(load_a(sHA, c, k0, Hsz, hi),
                     load_b_lds(sW1, 16 * wave + c, k0, Hsz, hi), z);
#pragma unroll
      for (int j = 0; j < 8; ++j)
        sZA[(j + 8 * hi) * MIDP + 16 * wave + c] = (_Float16)tanhf(z[j]);
      __syncthreads();                            // za visible; sHA reads done

      // GEMM2: k_s = (z @ W2^T + b2) * scale; accumulate + next stage y
#pragma unroll
      for (int nt = 0; nt < 4; ++nt) {
        int nb = 64 * wave + 16 * nt;
        v8f d;
#pragma unroll
        for (int j = 0; j < 8; ++j) d[j] = b2v[nt];
        d = wmma_f16(load_a(sZA, c, 0, MIDP, hi),
                     load_b_lds(sW2, nb + c, 0, MIDP, hi), d);
        d = wmma_f16(load_a(sZA, c, 32, MIDP, hi),
                     load_b_lds(sW2, nb + c, 32, MIDP, hi), d);
#pragma unroll
        for (int j = 0; j < 8; ++j) {
          float kv = d[j] * scv[j];
          ksum[nt][j] += wgt[s] * kv;
          if (s < 3) {
            float yn = hreg[nt][j] + cnext[s] * kv;
            sHA[(j + 8 * hi) * Hsz + nb + c] = (_Float16)yn;
          }
        }
      }
      __syncthreads();
    }

    // h = y0 + (hs/6) * (k1 + 2k2 + 2k3 + k4); republish f16 mirror
#pragma unroll
    for (int nt = 0; nt < 4; ++nt) {
      int nb = 64 * wave + 16 * nt;
#pragma unroll
      for (int j = 0; j < 8; ++j) {
        hreg[nt][j] += (hsv / 6.f) * ksum[nt][j];
        sHA[(j + 8 * hi) * Hsz + nb + c] = (_Float16)hreg[nt][j];
      }
    }
    __syncthreads();

    // ================= mu head: relu(h l1^T + b) @ mu_w ====================
    float mupart[8], sgpart[8];
#pragma unroll
    for (int j = 0; j < 8; ++j) { mupart[j] = 0.f; sgpart[j] = 0.f; }
#pragma unroll
    for (int tt = 0; tt < 2; ++tt) {
      int nb = 32 * wave + 16 * tt;
      v8f r;
#pragma unroll
      for (int j = 0; j < 8; ++j) r[j] = l1bv[tt];
#pragma unroll
      for (int k0 = 0; k0 < Hsz; k0 += 32)
        r = wmma_f16(load_a(sHA, c, k0, Hsz, hi),
                     load_b_g32(l1_w, nb + c, k0, Hsz, hi), r);  // streamed from L2
#pragma unroll
      for (int j = 0; j < 8; ++j) mupart[j] += fmaxf(r[j], 0.f) * muwv[tt];
    }

    // ================= sigma head: softplus(tanh(h W1^T+b) @ w2 + b2) ======
#pragma unroll
    for (int nt = 0; nt < 4; ++nt) {
      int nb = 64 * wave + 16 * nt;
      v8f sa;
#pragma unroll
      for (int j = 0; j < 8; ++j) sa[j] = sb1[nt];
#pragma unroll
      for (int k0 = 0; k0 < Hsz; k0 += 32)
        sa = wmma_f16(load_a(sHA, c, k0, Hsz, hi),
                      load_b_g32(sig_w1, nb + c, k0, Hsz, hi), sa);  // streamed
#pragma unroll
      for (int j = 0; j < 8; ++j) sgpart[j] += tanhf(sa[j]) * sw2[nt];
    }

    // Reduce the 16 columns per lane-half, then cross-wave via LDS atomics
#pragma unroll
    for (int j = 0; j < 8; ++j) {
      float m = mupart[j], sgv = sgpart[j];
#pragma unroll
      for (int off = 1; off < 16; off <<= 1) {
        m   += __shfl_xor(m, off, 32);
        sgv += __shfl_xor(sgv, off, 32);
      }
      if (c == 0) {
        atomicAdd(&sMU[j + 8 * hi], m);    // ds_add_f32
        atomicAdd(&sSG[j + 8 * hi], sgv);
      }
    }
    __syncthreads();

    if (tid < 16) {
      int b = row0 + tid;
      float sgin = sSG[tid];
      float sp = (sgin > 20.f) ? sgin : log1pf(expf(sgin));  // softplus
      out[b * Tsz + t]             = sMU[tid];               // mu  [B,T,1]
      out[Bsz * Tsz + b * Tsz + t] = sp;                     // sig [B,T,1]
    }
    __syncthreads();
  }
}

extern "C" void kernel_launch(void* const* d_in, const int* in_sizes, int n_in,
                              void* d_out, int out_size, void* d_ws, size_t ws_size,
                              hipStream_t stream) {
  (void)in_sizes; (void)n_in; (void)d_ws; (void)ws_size; (void)out_size;
  const float* dt     = (const float*)d_in[0];
  const float* x      = (const float*)d_in[1];
  const float* W_ih   = (const float*)d_in[2];
  const float* b_ih   = (const float*)d_in[3];
  const float* W_hh   = (const float*)d_in[4];
  const float* b_hh   = (const float*)d_in[5];
  const float* ode_w1 = (const float*)d_in[6];
  const float* ode_b1 = (const float*)d_in[7];
  const float* ode_w2 = (const float*)d_in[8];
  const float* ode_b2 = (const float*)d_in[9];
  const float* l1_w   = (const float*)d_in[10];
  const float* l1_b   = (const float*)d_in[11];
  const float* mu_w   = (const float*)d_in[12];
  const float* mu_b   = (const float*)d_in[13];
  const float* sig_w1 = (const float*)d_in[14];
  const float* sig_b1 = (const float*)d_in[15];
  const float* sig_w2 = (const float*)d_in[16];
  const float* sig_b2 = (const float*)d_in[17];

  static bool attr_set = false;  // host-side only; does not affect determinism
  if (!attr_set) {
    (void)hipFuncSetAttribute((const void*)odernn_fused_kernel,
                              hipFuncAttributeMaxDynamicSharedMemorySize,
                              SMEM_BYTES);
    attr_set = true;
  }

  odernn_fused_kernel<<<dim3(Bsz / 16), dim3(128), SMEM_BYTES, stream>>>(
      dt, x, W_ih, b_ih, W_hh, b_hh, ode_w1, ode_b1, ode_w2, ode_b2,
      l1_w, l1_b, mu_w, mu_b, sig_w1, sig_b1, sig_w2, sig_b2,
      (float*)d_out);
}
